// LocalSigLIPLoss_33457795235869
// MI455X (gfx1250) — matrix operand order repaired and checked by reading, not compiled, not run on hardware
//
#include <hip/hip_runtime.h>
#include <hip/hip_bf16.h>
#include <math.h>

typedef __attribute__((ext_vector_type(16))) _Float16 v16h;
typedef __attribute__((ext_vector_type(8)))  _Float16 h8;
typedef __attribute__((ext_vector_type(8)))  float    v8f;

#define B_SZ 8
#define T_SZ 1024
#define D_SZ 512
#define RT_N (T_SZ / 16)          /* 64 row tiles per batch */
#define LOSS_BLOCKS ((B_SZ * T_SZ) / 256)   /* 32 */

#define CAT16(lo, hi) __builtin_shufflevector((lo), (hi), 0,1,2,3,4,5,6,7,8,9,10,11,12,13,14,15)

// ---------------------------------------------------------------------------
// Pass 1: L2-normalize rows (fp32) -> f16, one wave per row (32 lanes x 16 elems)
// ---------------------------------------------------------------------------
__global__ void __launch_bounds__(256)
siglip_normalize_kernel(const float* __restrict__ motion,
                        const float* __restrict__ audio,
                        _Float16* __restrict__ m_h,
                        _Float16* __restrict__ a_h) {
    const int gtid = blockIdx.x * blockDim.x + threadIdx.x;
    const int wave = gtid >> 5;            // 0 .. 2*B*T-1
    const int lane = threadIdx.x & 31;
    const int ROWS = B_SZ * T_SZ;

    const float* src;
    _Float16* dst;
    int row;
    if (wave < ROWS) { src = motion; dst = m_h; row = wave; }
    else             { src = audio;  dst = a_h; row = wave - ROWS; }

    const float4* p = (const float4*)(src + (size_t)row * D_SZ + lane * 16);
    float4 x0 = p[0], x1 = p[1], x2 = p[2], x3 = p[3];

    float ss = x0.x*x0.x + x0.y*x0.y + x0.z*x0.z + x0.w*x0.w
             + x1.x*x1.x + x1.y*x1.y + x1.z*x1.z + x1.w*x1.w
             + x2.x*x2.x + x2.y*x2.y + x2.z*x2.z + x2.w*x2.w
             + x3.x*x3.x + x3.y*x3.y + x3.z*x3.z + x3.w*x3.w;
    // wave32 butterfly reduction
    #pragma unroll
    for (int m = 16; m >= 1; m >>= 1) ss += __shfl_xor(ss, m, 32);

    const float inv = 1.0f / fmaxf(sqrtf(ss), 1e-12f);

    h8 o0, o1;
    o0[0] = (_Float16)(x0.x * inv); o0[1] = (_Float16)(x0.y * inv);
    o0[2] = (_Float16)(x0.z * inv); o0[3] = (_Float16)(x0.w * inv);
    o0[4] = (_Float16)(x1.x * inv); o0[5] = (_Float16)(x1.y * inv);
    o0[6] = (_Float16)(x1.z * inv); o0[7] = (_Float16)(x1.w * inv);
    o1[0] = (_Float16)(x2.x * inv); o1[1] = (_Float16)(x2.y * inv);
    o1[2] = (_Float16)(x2.z * inv); o1[3] = (_Float16)(x2.w * inv);
    o1[4] = (_Float16)(x3.x * inv); o1[5] = (_Float16)(x3.y * inv);
    o1[6] = (_Float16)(x3.z * inv); o1[7] = (_Float16)(x3.w * inv);

    h8* q = (h8*)(dst + (size_t)row * D_SZ + lane * 16);
    q[0] = o0;
    q[1] = o1;
}

// ---------------------------------------------------------------------------
// Pass 2: banded GEMM S = m_norm * a_norm^T, band half-width 16.
// One wave per (b, rowtile, coltile(0..2)); 16 chained v_wmma_f32_16x16x32_f16.
// Output tile layout: W[b][rt][ct][M][N] (16x16 f32).
// ---------------------------------------------------------------------------
__global__ void __launch_bounds__(32)
siglip_band_wmma_kernel(const _Float16* __restrict__ m_h,
                        const _Float16* __restrict__ a_h,
                        float* __restrict__ W) {
    const int ct = blockIdx.x;             // 0..2
    const int rt = blockIdx.y;             // 0..63
    const int b  = blockIdx.z;             // 0..7
    const int c0 = rt * 16 + (ct - 1) * 16;
    if (c0 < 0 || c0 >= T_SZ) return;      // whole-tile OOB (uniform branch)
    const int r0 = rt * 16;

    const int lane = threadIdx.x;
    const int half = lane >> 4;            // 0 or 1
    const int l15  = lane & 15;

    // A fragment (16x32, f16): lane half 0 -> K {0..7,16..23}; half 1 -> {8..15,24..31}
    const _Float16* aBase = m_h + ((size_t)(b * T_SZ + r0 + l15)) * D_SZ + half * 8;
    // B fragment (32x16, f16): lane half 0 -> K {0..7,16..23}; half 1 -> {8..15,24..31}
    const _Float16* bBase = a_h + ((size_t)(b * T_SZ + c0 + l15)) * D_SZ + half * 8;

    v8f acc = {};
    #pragma unroll 4
    for (int kk = 0; kk < D_SZ; kk += 32) {
        h8 alo = *(const h8*)(aBase + kk);
        h8 ahi = *(const h8*)(aBase + kk + 16);
        h8 blo = *(const h8*)(bBase + kk);
        h8 bhi = *(const h8*)(bBase + kk + 16);
        v16h Af = CAT16(alo, ahi);
        v16h Bf = CAT16(blo, bhi);
        acc = __builtin_amdgcn_wmma_f32_16x16x32_f16(
            /*neg_a=*/false, Af, /*neg_b=*/false, Bf,
            /*c_mod=*/(short)0, acc, /*reuse_a=*/false, /*reuse_b=*/false);
    }

    // D layout: acc[v] = (M = v + 8*half, N = l15)
    float* out = W + ((((size_t)b * RT_N + rt) * 3 + ct) << 8);
    #pragma unroll
    for (int v = 0; v < 8; ++v)
        out[(v + half * 8) * 16 + l15] = acc[v];
}

// ---------------------------------------------------------------------------
// Pass 3: SigLIP loss + argmax accuracy over the band, both directions.
// One thread per (b,t); deterministic block reduction into partials.
// ---------------------------------------------------------------------------
__device__ __forceinline__ float band_read(const float* __restrict__ W,
                                           int b, int row, int col) {
    const int rt = row >> 4;
    const int q  = col - rt * 16 + 16;     // 0..47 guaranteed by band width
    return W[((((size_t)b * RT_N + rt) * 3 + (q >> 4)) << 8) + ((row & 15) << 4) + (q & 15)];
}

__device__ __forceinline__ float log_sigmoid_f(float x) {
    return fminf(x, 0.0f) - log1pf(expf(-fabsf(x)));
}

__global__ void __launch_bounds__(256)
siglip_loss_kernel(const float* __restrict__ W,
                   const float* __restrict__ scale_p,
                   const float* __restrict__ bias_p,
                   float* __restrict__ partials) {
    const int gid = blockIdx.x * blockDim.x + threadIdx.x;   // 0..B*T-1
    const int b = gid >> 10;
    const int t = gid & (T_SZ - 1);
    const float scale = expf(scale_p[0]);
    const float bias  = bias_p[0];

    const int start = max(0, t - 4);
    const int end   = min(T_SZ, t + 4);
    const int ns0   = max(0, start - 12);
    const int ne1   = min(T_SZ, end + 12);

    float loss = 0.0f, corr = 0.0f;

    #pragma unroll
    for (int dir = 0; dir < 2; ++dir) {
        float best = -1e30f;
        float bestPos = 0.0f;
        // positives first (argmax tie-break = first occurrence, strict >)
        for (int j = start; j < end; ++j) {
            float raw = (dir == 0) ? band_read(W, b, t, j) : band_read(W, b, j, t);
            float L = raw * scale + bias;
            loss -= log_sigmoid_f(L);
            if (L > best) { best = L; bestPos = 1.0f; }
        }
        for (int j = ns0; j < start; ++j) {
            float raw = (dir == 0) ? band_read(W, b, t, j) : band_read(W, b, j, t);
            float L = raw * scale + bias;
            loss -= log_sigmoid_f(-L);
            if (L > best) { best = L; bestPos = 0.0f; }
        }
        for (int j = end; j < ne1; ++j) {
            float raw = (dir == 0) ? band_read(W, b, t, j) : band_read(W, b, j, t);
            float L = raw * scale + bias;
            loss -= log_sigmoid_f(-L);
            if (L > best) { best = L; bestPos = 0.0f; }
        }
        corr += bestPos;
    }

    __shared__ float sl[256];
    __shared__ float sc[256];
    const int tid = threadIdx.x;
    sl[tid] = loss;
    sc[tid] = corr;
    __syncthreads();
    #pragma unroll
    for (int s = 128; s >= 1; s >>= 1) {
        if (tid < s) { sl[tid] += sl[tid + s]; sc[tid] += sc[tid + s]; }
        __syncthreads();
    }
    if (tid == 0) {
        partials[2 * blockIdx.x + 0] = sl[0];
        partials[2 * blockIdx.x + 1] = sc[0];
    }
}

__global__ void siglip_finalize_kernel(const float* __restrict__ partials,
                                       float* __restrict__ out) {
    if (blockIdx.x == 0 && threadIdx.x == 0) {
        float l = 0.0f, c = 0.0f;
        for (int i = 0; i < LOSS_BLOCKS; ++i) {   // fixed order: deterministic
            l += partials[2 * i + 0];
            c += partials[2 * i + 1];
        }
        out[0] = l / (float)(B_SZ * 2 * T_SZ);    // (1/B sum) / (2T)
        out[1] = c / (float)(2 * T_SZ * B_SZ);
    }
}

// ---------------------------------------------------------------------------
extern "C" void kernel_launch(void* const* d_in, const int* in_sizes, int n_in,
                              void* d_out, int out_size, void* d_ws, size_t ws_size,
                              hipStream_t stream) {
    (void)in_sizes; (void)n_in; (void)out_size; (void)ws_size;

    const float* motion  = (const float*)d_in[0];
    const float* audio   = (const float*)d_in[1];
    const float* lscale  = (const float*)d_in[2];
    const float* lbias   = (const float*)d_in[3];
    float* out = (float*)d_out;

    const size_t MH_BYTES = (size_t)B_SZ * T_SZ * D_SZ * 2;        // 8 MB each
    const size_t W_BYTES  = (size_t)B_SZ * RT_N * 3 * 256 * 4;     // 1.5 MB

    char* ws = (char*)d_ws;
    _Float16* m_h = (_Float16*)ws;
    _Float16* a_h = (_Float16*)(ws + MH_BYTES);
    float*    Wb  = (float*)(ws + 2 * MH_BYTES);
    float*    partials = (float*)(ws + 2 * MH_BYTES + W_BYTES);

    // Pass 1: normalize both tensors to f16 (2*B*T rows, 1 wave/row, 8 waves/block)
    siglip_normalize_kernel<<<(2 * B_SZ * T_SZ) / 8, 256, 0, stream>>>(motion, audio, m_h, a_h);

    // Pass 2: banded WMMA GEMM (1 wave per tile)
    dim3 bandGrid(3, RT_N, B_SZ);
    siglip_band_wmma_kernel<<<bandGrid, 32, 0, stream>>>(m_h, a_h, Wb);

    // Pass 3: loss + accuracy, block partials
    siglip_loss_kernel<<<LOSS_BLOCKS, 256, 0, stream>>>(Wb, lscale, lbias, partials);

    // Pass 4: deterministic final reduction
    siglip_finalize_kernel<<<1, 32, 0, stream>>>(partials, out);
}
